// PaiNN_18107582120333
// MI455X (gfx1250) — matrix operand order repaired and checked by reading, not compiled, not run on hardware
//
#include <hip/hip_runtime.h>
#include <hip/hip_bf16.h>

typedef __attribute__((ext_vector_type(16))) _Float16 v16h;
typedef __attribute__((ext_vector_type(8)))  _Float16 v8h;
typedef __attribute__((ext_vector_type(8)))  float    v8f;
typedef __attribute__((ext_vector_type(4)))  float    v4f;

#define N_NODES 10000
#define N_EDGES 160000
#define DIM     128
#define NB      20
#define NLAYER  3
#define PI_F    3.14159265358979323846f

__device__ __forceinline__ float silu_f(float x) {
    return x * __builtin_amdgcn_rcpf(1.0f + __expf(-x));
}

// ---------------------------------------------------------------------------
// Weight pre-pack: W[K,Nc] f32 (row-major) -> Wt[Nc,K] f16 (transposed).
// Runs once per weight per launch; makes GEMM B-staging a pure f16 copy.
// ---------------------------------------------------------------------------
__global__ void pack_weight_kernel(const float* __restrict__ W,
                                   _Float16* __restrict__ Wt,
                                   int K, int Nc) {
    int idx = blockIdx.x * blockDim.x + threadIdx.x;
    if (idx >= K * Nc) return;
    int k = idx / Nc, c = idx - k * Nc;
    Wt[(size_t)c * K + k] = (_Float16)W[idx];
}

// ---------------------------------------------------------------------------
// WMMA GEMM: out[M,Nc] = act(A[M,K] @ W[K,Nc] (+ bias)), W pre-packed as
// Wt[Nc,K] f16. Block = 128 threads (4 waves), tile 64(M) x 64(N), K-step 32.
// Wave w owns rows w*16..+15, all 64 cols -> 4 accumulators.
// A staged f32->f16 in LDS (coalesced b128); B tile copied f16->LDS
// (contiguous b128 both ways). K % 32 == 0, Nc % 64 == 0; M clamp+predicate.
// ---------------------------------------------------------------------------
template<int ACT, bool HAS_BIAS>
__global__ __launch_bounds__(128)
void gemm_wmma(const float* __restrict__ A, const _Float16* __restrict__ Wt,
               const float* __restrict__ bias, float* __restrict__ out,
               int M, int K, int Nc) {
    __shared__ _Float16 Alds[64 * 32];     // [row][k]
    __shared__ _Float16 Blds[64 * 32];     // [col][k] (already transposed)

    const int tid  = threadIdx.x;
    const int lane = tid & 31;
    const int wave = tid >> 5;
    const int grp  = lane >> 4;
    const int lid  = lane & 15;

    const int rowBlock = blockIdx.y * 64;
    const int colBase  = blockIdx.x * 64;

    // staging split (128 threads)
    const int a_row = tid >> 1;            // 0..63
    const int a_seg = (tid & 1) * 16;      // 0 / 16
    const int b_col = tid >> 1;            // 0..63
    const int b_seg = (tid & 1) * 16;      // 0 / 16

    const int a_grow = min(rowBlock + a_row, M - 1);   // clamp: EXEC stays full
    const float*    aptr = A  + (size_t)a_grow * K + a_seg;
    const _Float16* bptr = Wt + (size_t)(colBase + b_col) * K + b_seg;

    v8f acc[4] = {{}, {}, {}, {}};

    for (int k0 = 0; k0 < K; k0 += 32) {
        // ---- stage A tile (64x32 f32 -> f16, coalesced) ----
        const float* ap = aptr + k0;
        v4f av0 = *(const v4f*)(ap + 0);
        v4f av1 = *(const v4f*)(ap + 4);
        v4f av2 = *(const v4f*)(ap + 8);
        v4f av3 = *(const v4f*)(ap + 12);
        v8h ah0, ah1;
        #pragma unroll
        for (int e = 0; e < 4; ++e) {
            ah0[e]     = (_Float16)av0[e];
            ah0[e + 4] = (_Float16)av1[e];
            ah1[e]     = (_Float16)av2[e];
            ah1[e + 4] = (_Float16)av3[e];
        }
        _Float16* as = Alds + a_row * 32 + a_seg;
        *(v8h*)(as)     = ah0;
        *(v8h*)(as + 8) = ah1;

        // ---- stage B tile (64 cols x 32 k): pure f16 copy, b128 in/out ----
        const _Float16* bp = bptr + k0;
        v8h bh0 = *(const v8h*)(bp);
        v8h bh1 = *(const v8h*)(bp + 8);
        _Float16* bs = Blds + b_col * 32 + b_seg;
        *(v8h*)(bs)     = bh0;
        *(v8h*)(bs + 8) = bh1;

        // prefetch next K tiles
        if (k0 + 32 < K) {
            __builtin_prefetch(ap + 32, 0, 0);
            __builtin_prefetch(bp + 32, 0, 0);
        }
        __syncthreads();

        // ---- A fragment: elements 0-7 <-> k=grp*8..+7, 8-15 <-> 16+grp*8..+7
        const _Float16* arow = Alds + (wave * 16 + lid) * 32 + grp * 8;
        v8h alo = *(const v8h*)(arow);
        v8h ahi = *(const v8h*)(arow + 16);
        v16h af = __builtin_shufflevector(alo, ahi,
                   0, 1, 2, 3, 4, 5, 6, 7, 8, 9, 10, 11, 12, 13, 14, 15);

        // ---- 4 column sub-tiles ----
        #pragma unroll
        for (int c = 0; c < 4; ++c) {
            const _Float16* bcol = Blds + (c * 16 + lid) * 32 + grp * 8;
            v16h bf = __builtin_shufflevector(*(const v8h*)(bcol), *(const v8h*)(bcol + 16),
                   0, 1, 2, 3, 4, 5, 6, 7, 8, 9, 10, 11, 12, 13, 14, 15);
            acc[c] = __builtin_amdgcn_wmma_f32_16x16x32_f16(
                         false, af, false, bf, (short)0, acc[c], false, false);
        }
        __syncthreads();
    }

    // ---- epilogue: bias + activation + predicated store ----
    #pragma unroll
    for (int c = 0; c < 4; ++c) {
        const int col = colBase + c * 16 + lid;
        float bv = 0.0f;
        if (HAS_BIAS) bv = bias[col];
        #pragma unroll
        for (int v = 0; v < 8; ++v) {
            int r = rowBlock + wave * 16 + v + 8 * grp;
            if (r < M) {
                float x = acc[c][v] + bv;
                if (ACT == 1) x = silu_f(x);
                out[(size_t)r * Nc + col] = x;
            }
        }
    }
}

// ---------------------------------------------------------------------------
__global__ void edge_geom_kernel(const float* __restrict__ pos,
                                 const int* __restrict__ ei,
                                 const int* __restrict__ ej,
                                 float* __restrict__ rbf,
                                 float* __restrict__ rsh,
                                 float* __restrict__ env) {
    int e = blockIdx.x * blockDim.x + threadIdx.x;
    if (e >= N_EDGES) return;
    int i = ei[e], j = ej[e];
    float vx = pos[j * 3 + 0] - pos[i * 3 + 0];
    float vy = pos[j * 3 + 1] - pos[i * 3 + 1];
    float vz = pos[j * 3 + 2] - pos[i * 3 + 2];
    float r  = sqrtf(vx * vx + vy * vy + vz * vz + 1e-12f);
    float inv = __builtin_amdgcn_rcpf(r);
    rsh[e * 3 + 0] = vx * inv;
    rsh[e * 3 + 1] = vy * inv;
    rsh[e * 3 + 2] = vz * inv;
    const float coef = 0.6324555320336759f;       // sqrt(2/5)
    float arg = PI_F * r * 0.2f;                  // pi*r/cutoff
    #pragma unroll 4
    for (int nb = 0; nb < NB; ++nb)
        rbf[(size_t)e * NB + nb] = coef * __sinf((float)(nb + 1) * arg) * inv;
    env[e] = 0.5f * (__cosf(arg) + 1.0f) * (r < 5.0f ? 1.0f : 0.0f);
}

// ---------------------------------------------------------------------------
__global__ void embed_kernel(const int* __restrict__ at_no,
                             const float* __restrict__ table,
                             float* __restrict__ xs) {
    int idx = blockIdx.x * blockDim.x + threadIdx.x;
    if (idx >= N_NODES * DIM) return;
    int n = idx >> 7, d = idx & 127;
    xs[idx] = table[(size_t)at_no[n] * DIM + d];
}

__global__ void zero_kernel(float* __restrict__ p, int count) {
    int idx = blockIdx.x * blockDim.x + threadIdx.x;
    if (idx < count) p[idx] = 0.0f;
}

__global__ void copy_kernel(float* __restrict__ dst, const float* __restrict__ src,
                            int count) {
    int idx = blockIdx.x * blockDim.x + threadIdx.x;
    if (idx < count) dst[idx] = src[idx];
}

// ---------------------------------------------------------------------------
// Message + scatter. One block (128 threads) per edge; rbf/dir staged in LDS.
// filt = (rbf @ Wr + br) * env computed in-register (K=20, L2-resident).
// ---------------------------------------------------------------------------
__global__ __launch_bounds__(128)
void message_kernel(const int* __restrict__ ei, const int* __restrict__ ej,
                    const float* __restrict__ phi,
                    const float* __restrict__ rbf,
                    const float* __restrict__ rsh,
                    const float* __restrict__ env,
                    const float* __restrict__ Wr,   // [NB, 3*DIM]
                    const float* __restrict__ br,   // [3*DIM]
                    const float* __restrict__ xv_old,
                    float* __restrict__ xv_new,
                    float* __restrict__ xs) {
    __shared__ float srbf[NB];
    __shared__ float sdir[3];
    const int e = blockIdx.x;
    const int d = threadIdx.x;      // 0..127
    if (d < NB)            srbf[d] = rbf[(size_t)e * NB + d];
    if (d >= 32 && d < 35) sdir[d - 32] = rsh[e * 3 + (d - 32)];
    __syncthreads();

    const int i = ei[e], j = ej[e];
    const float en = env[e];

    float f0 = br[d], f1 = br[DIM + d], f2 = br[2 * DIM + d];
    #pragma unroll 5
    for (int k = 0; k < NB; ++k) {
        const float rb = srbf[k];
        const float* wrow = Wr + (size_t)k * (3 * DIM);
        f0 += rb * wrow[d];
        f1 += rb * wrow[DIM + d];
        f2 += rb * wrow[2 * DIM + d];
    }
    f0 *= en; f1 *= en; f2 *= en;

    const float* pj = phi + (size_t)j * (3 * DIM);
    const float m_s  = pj[d]           * f0;
    const float m_vv = pj[DIM + d]     * f1;
    const float m_vs = pj[2 * DIM + d] * f2;

    atomicAdd(&xs[(size_t)i * DIM + d], m_s);
    #pragma unroll
    for (int c = 0; c < 3; ++c) {
        float v = xv_old[((size_t)j * 3 + c) * DIM + d] * m_vv + sdir[c] * m_vs;
        atomicAdd(&xv_new[((size_t)i * 3 + c) * DIM + d], v);
    }
}

// ---------------------------------------------------------------------------
__global__ void vn_cat_kernel(const float* __restrict__ xs,
                              const float* __restrict__ Vv,
                              float* __restrict__ cat) {
    int idx = blockIdx.x * blockDim.x + threadIdx.x;
    if (idx >= N_NODES * DIM) return;
    int n = idx >> 7, d = idx & 127;
    float s = 1e-8f;
    #pragma unroll
    for (int c = 0; c < 3; ++c) {
        float v = Vv[((size_t)n * 3 + c) * DIM + d];
        s += v * v;
    }
    cat[(size_t)n * (2 * DIM) + d]       = xs[idx];
    cat[(size_t)n * (2 * DIM) + DIM + d] = sqrtf(s);
}

// ---------------------------------------------------------------------------
__global__ void update_kernel(float* __restrict__ xs, float* __restrict__ xv,
                              const float* __restrict__ Uv,
                              const float* __restrict__ Vv,
                              const float* __restrict__ a) {
    int idx = blockIdx.x * blockDim.x + threadIdx.x;
    if (idx >= N_NODES * DIM) return;
    int n = idx >> 7, d = idx & 127;
    const float* an = a + (size_t)n * (3 * DIM);
    float a_vv = an[d], a_sv = an[DIM + d], a_ss = an[2 * DIM + d];
    float dot = 0.0f;
    #pragma unroll
    for (int c = 0; c < 3; ++c) {
        size_t o = ((size_t)n * 3 + c) * DIM + d;
        float u = Uv[o], v = Vv[o];
        dot += u * v;
        xv[o] += a_vv * u;
    }
    xs[idx] += a_ss + a_sv * dot;
}

// ---------------------------------------------------------------------------
__global__ void head_reduce_kernel(const float* __restrict__ h,  // [N, 64]
                                   const float* __restrict__ W2, // [64]
                                   const float* __restrict__ b2, // [1]
                                   float* __restrict__ out) {
    int n = blockIdx.x * blockDim.x + threadIdx.x;
    float s = 0.0f;
    if (n < N_NODES) {
        s = b2[0];
        #pragma unroll 8
        for (int d = 0; d < DIM / 2; ++d)
            s += h[(size_t)n * (DIM / 2) + d] * W2[d];
    }
    #pragma unroll
    for (int off = 16; off > 0; off >>= 1)
        s += __shfl_down(s, off, 32);
    if ((threadIdx.x & 31) == 0)
        atomicAdd(out, s);
}

// ---------------------------------------------------------------------------
extern "C" void kernel_launch(void* const* d_in, const int* in_sizes, int n_in,
                              void* d_out, int out_size, void* d_ws, size_t ws_size,
                              hipStream_t stream) {
    const float* pos     = (const float*)d_in[0];
    const int*   eidx    = (const int*)  d_in[1];
    const int*   at_no   = (const int*)  d_in[2];
    const float* table   = (const float*)d_in[3];
    const float* msg_W1  = (const float*)d_in[4];
    const float* msg_b1  = (const float*)d_in[5];
    const float* msg_W2  = (const float*)d_in[6];
    const float* msg_b2  = (const float*)d_in[7];
    const float* msg_Wr  = (const float*)d_in[8];
    const float* msg_br  = (const float*)d_in[9];
    const float* upd_U   = (const float*)d_in[10];
    const float* upd_V   = (const float*)d_in[11];
    const float* upd_W1  = (const float*)d_in[12];
    const float* upd_b1  = (const float*)d_in[13];
    const float* upd_W2  = (const float*)d_in[14];
    const float* upd_b2  = (const float*)d_in[15];
    const float* out_W1  = (const float*)d_in[16];
    const float* out_b1  = (const float*)d_in[17];
    const float* out_W2  = (const float*)d_in[18];
    const float* out_b2  = (const float*)d_in[19];

    const int* ei = eidx;             // receivers (row 0)
    const int* ej = eidx + N_EDGES;   // senders   (row 1)

    // ---- workspace carve-up ----
    float* w = (float*)d_ws;
    float* rbf = w;  w += (size_t)N_EDGES * NB;
    float* rsh = w;  w += (size_t)N_EDGES * 3;
    float* env = w;  w += (size_t)N_EDGES;
    float* xs  = w;  w += (size_t)N_NODES * DIM;
    float* xvA = w;  w += (size_t)N_NODES * 3 * DIM;
    float* xvB = w;  w += (size_t)N_NODES * 3 * DIM;
    float* h1  = w;  w += (size_t)N_NODES * DIM;        // also head hidden [N,64]
    float* phi = w;  w += (size_t)N_NODES * 3 * DIM;    // also gate vector a
    float* Uv  = w;  w += (size_t)N_NODES * 3 * DIM;
    float* Vv  = w;  w += (size_t)N_NODES * 3 * DIM;
    float* cat = w;  w += (size_t)N_NODES * 2 * DIM;
    // f16 packed (transposed) weights, reused across layers
    _Float16* hw = (_Float16*)w;
    _Float16* pW1  = hw;  hw += DIM * DIM;              // [128][128]
    _Float16* pW2  = hw;  hw += DIM * 3 * DIM;          // [384][128]
    _Float16* pU   = hw;  hw += DIM * DIM;
    _Float16* pV   = hw;  hw += DIM * DIM;
    _Float16* puW1 = hw;  hw += 2 * DIM * DIM;          // [128][256]
    _Float16* puW2 = hw;  hw += DIM * 3 * DIM;
    _Float16* poW1 = hw;  hw += DIM * (DIM / 2);        // [64][128]

    const int T = 256;
    const int ND  = N_NODES * DIM;
    const int NV3 = N_NODES * 3 * DIM;

    auto tiles = [](int m) { return (m + 63) / 64; };
    auto packg = [](int n) { return (n + 255) / 256; };

    // ---- geometry + init ----
    edge_geom_kernel<<<(N_EDGES + T - 1) / T, T, 0, stream>>>(pos, ei, ej, rbf, rsh, env);
    embed_kernel<<<(ND + T - 1) / T, T, 0, stream>>>(at_no, table, xs);
    zero_kernel<<<(NV3 + T - 1) / T, T, 0, stream>>>(xvA, NV3);
    pack_weight_kernel<<<packg(DIM * DIM / 2), T, 0, stream>>>(out_W1, poW1, DIM, DIM / 2);

    float* xv_cur = xvA;
    float* xv_alt = xvB;
    const dim3 blk(128);

    for (int l = 0; l < NLAYER; ++l) {
        const float* W1 = msg_W1 + (size_t)l * DIM * DIM;
        const float* b1 = msg_b1 + (size_t)l * DIM;
        const float* W2 = msg_W2 + (size_t)l * DIM * 3 * DIM;
        const float* b2 = msg_b2 + (size_t)l * 3 * DIM;
        const float* Wr = msg_Wr + (size_t)l * NB * 3 * DIM;
        const float* br = msg_br + (size_t)l * 3 * DIM;
        const float* Ul = upd_U  + (size_t)l * DIM * DIM;
        const float* Vl = upd_V  + (size_t)l * DIM * DIM;
        const float* uW1 = upd_W1 + (size_t)l * 2 * DIM * DIM;
        const float* ub1 = upd_b1 + (size_t)l * DIM;
        const float* uW2 = upd_W2 + (size_t)l * DIM * 3 * DIM;
        const float* ub2 = upd_b2 + (size_t)l * 3 * DIM;

        // pack this layer's weights to f16 transposed
        pack_weight_kernel<<<packg(DIM * DIM),     T, 0, stream>>>(W1,  pW1,  DIM,     DIM);
        pack_weight_kernel<<<packg(DIM * 3 * DIM), T, 0, stream>>>(W2,  pW2,  DIM,     3 * DIM);
        pack_weight_kernel<<<packg(DIM * DIM),     T, 0, stream>>>(Ul,  pU,   DIM,     DIM);
        pack_weight_kernel<<<packg(DIM * DIM),     T, 0, stream>>>(Vl,  pV,   DIM,     DIM);
        pack_weight_kernel<<<packg(2 * DIM * DIM), T, 0, stream>>>(uW1, puW1, 2 * DIM, DIM);
        pack_weight_kernel<<<packg(DIM * 3 * DIM), T, 0, stream>>>(uW2, puW2, DIM,     3 * DIM);

        // phi = silu(xs@W1+b1) @ W2 + b2
        gemm_wmma<1, true><<<dim3(DIM / 64, tiles(N_NODES)), blk, 0, stream>>>(
            xs, pW1, b1, h1, N_NODES, DIM, DIM);
        gemm_wmma<0, true><<<dim3(3 * DIM / 64, tiles(N_NODES)), blk, 0, stream>>>(
            h1, pW2, b2, phi, N_NODES, DIM, 3 * DIM);

        // scatter messages: xv double buffer preserves "old xv" reads
        copy_kernel<<<(NV3 + T - 1) / T, T, 0, stream>>>(xv_alt, xv_cur, NV3);
        message_kernel<<<N_EDGES, 128, 0, stream>>>(
            ei, ej, phi, rbf, rsh, env, Wr, br, xv_cur, xv_alt, xs);
        { float* t = xv_cur; xv_cur = xv_alt; xv_alt = t; }

        // update: Uv = xv@U, Vv = xv@V  (xv viewed as [3N, DIM])
        gemm_wmma<0, false><<<dim3(DIM / 64, tiles(3 * N_NODES)), blk, 0, stream>>>(
            xv_cur, pU, nullptr, Uv, 3 * N_NODES, DIM, DIM);
        gemm_wmma<0, false><<<dim3(DIM / 64, tiles(3 * N_NODES)), blk, 0, stream>>>(
            xv_cur, pV, nullptr, Vv, 3 * N_NODES, DIM, DIM);

        vn_cat_kernel<<<(ND + T - 1) / T, T, 0, stream>>>(xs, Vv, cat);
        gemm_wmma<1, true><<<dim3(DIM / 64, tiles(N_NODES)), blk, 0, stream>>>(
            cat, puW1, ub1, h1, N_NODES, 2 * DIM, DIM);
        gemm_wmma<0, true><<<dim3(3 * DIM / 64, tiles(N_NODES)), blk, 0, stream>>>(
            h1, puW2, ub2, phi, N_NODES, DIM, 3 * DIM);

        update_kernel<<<(ND + T - 1) / T, T, 0, stream>>>(xs, xv_cur, Uv, Vv, phi);
    }

    // ---- output head: e = silu(xs@oW1+ob1)@oW2+ob2, summed over nodes ----
    gemm_wmma<1, true><<<dim3((DIM / 2) / 64, tiles(N_NODES)), blk, 0, stream>>>(
        xs, poW1, out_b1, h1, N_NODES, DIM, DIM / 2);
    zero_kernel<<<1, 32, 0, stream>>>((float*)d_out, 1);
    head_reduce_kernel<<<(N_NODES + T - 1) / T, T, 0, stream>>>(
        h1, out_W2, out_b2, (float*)d_out);
}
